// AttentionModel_26551487823959
// MI455X (gfx1250) — matrix operand order repaired and checked by reading, not compiled, not run on hardware
//
#include <hip/hip_runtime.h>
#include <hip/hip_bf16.h>
#include <stdint.h>
#include <math.h>

// ---------------------------------------------------------------------------
// AttentionModel greedy decode for MI455X (gfx1250, wave32).
//  - Precompute GEMM (128000x128 @ 128x384) via v_wmma_f32_16x16x32_bf16,
//    with W_node staged transposed in LDS as bf16 (ds_load_b128 B-fragments)
//    and A fragments loaded once per wave and reused across column tiles.
//  - gK/gV/lK stored bf16 (98 MB -> fits 192 MB L2) to halve per-step traffic.
//  - 64 sequential decode steps: attention scan + logits scan are memory-bound
//    batched GEMVs -> VALU with bit-trick bf16 decode + global prefetch.
// ---------------------------------------------------------------------------

typedef __attribute__((ext_vector_type(16))) __bf16 v16bf;
typedef __attribute__((ext_vector_type(8)))  float  v8f;

#define B_    128
#define N_    1000
#define D_    128
#define H_    8
#define DK_   16
#define COLS_ 384   // 3*D
#define JH_   192   // columns staged per block (half of COLS_)

__device__ __forceinline__ uint16_t f32_to_bf16_bits(float f) {
  uint32_t u = __float_as_uint(f);
  return (uint16_t)((u + 0x7FFFu + ((u >> 16) & 1u)) >> 16);  // RNE
}
__device__ __forceinline__ __bf16 f2bf(float f) {
  uint16_t b = f32_to_bf16_bits(f);
  __bf16 h;
  __builtin_memcpy(&h, &b, 2);
  return h;
}

// ---------------------------------------------------------------------------
// Precompute: kvq = emb @ W_node, scattered into bf16 gK/gV (B,H,N,16),
// lK (B,N,128).  Grid = (8000 M-tiles, 2 column halves), 128 threads = 4 waves.
// Each wave: A fragments (4 K-chunks) loaded once, then 3 column tiles of the
// LDS-staged, transposed bf16 W_node half -> 12 WMMAs per wave.
// ---------------------------------------------------------------------------
__global__ void __launch_bounds__(128)
precompute_gemm(const float* __restrict__ emb, const float* __restrict__ Wn,
                uint16_t* __restrict__ gK, uint16_t* __restrict__ gV,
                uint16_t* __restrict__ lK) {
  __shared__ uint16_t wnT[JH_ * D_];            // [jloc][k] bf16, 48 KB
  const int tid  = threadIdx.x;
  const int mt   = blockIdx.x;                  // M tile: 16 rows of B*N
  const int half = blockIdx.y;                  // which 192-column half
  const int jbase = half * JH_;

  // Stage this half of W_node into LDS, transposed, bf16.  Coalesced reads.
  for (int idx = tid; idx < D_ * JH_; idx += 128) {
    int k = idx / JH_;
    int jl = idx - k * JH_;
    wnT[jl * D_ + k] = f32_to_bf16_bits(Wn[(size_t)k * COLS_ + jbase + jl]);
  }
  __syncthreads();

  const int lane = tid & 31;
  const int wave = tid >> 5;
  const int l16  = lane & 15;
  const int bh   = lane >> 4;                   // half-wave select
  const int mrow = mt * 16 + l16;               // A-matrix row (M = lane%16)

  // Load + convert A fragments for all 4 K32 chunks (contiguous 32B chunks).
  v16bf a[4];
  #pragma unroll
  for (int kc = 0; kc < 4; ++kc) {
    float lo[8], hi[8];
    __builtin_memcpy(lo, emb + (size_t)mrow * D_ + kc * 32 + 16 * bh, 32);
    __builtin_memcpy(hi, emb + (size_t)mrow * D_ + kc * 32 + 16 * bh + 8, 32);
    #pragma unroll
    for (int e = 0; e < 8; ++e) {
      a[kc][e]     = f2bf(lo[e]);               // lanes0-15 K=e,  lanes16-31 +16
      a[kc][e + 8] = f2bf(hi[e]);               // lanes0-15 K=8+e, ...
    }
  }

  // 3 column tiles per wave (12 tiles per half).
  #pragma unroll
  for (int t = 0; t < 3; ++t) {
    const int jtl = wave * 3 + t;               // local column tile
    // B 32x16 bf16 layout: N = lane%16, K = e + 16*(lane>=16)
    const uint16_t* col = wnT + (size_t)(jtl * 16 + l16) * D_;
    v8f c = {};
    #pragma unroll
    for (int kc = 0; kc < 4; ++kc) {
      v16bf bm;
      __builtin_memcpy(&bm, col + kc * 32 + 16 * bh, 32);  // 2x ds_load_b128
      c = __builtin_amdgcn_wmma_f32_16x16x32_bf16(false, a[kc], false, bm,
                                                  (short)0, c, false, false);
    }
    // C/D layout: lane l, vgpr v -> N = l%16, M = v + 8*(l>=16)
    #pragma unroll
    for (int v = 0; v < 8; ++v) {
      int gm = mt * 16 + v + 8 * bh;            // global row in [0, B*N)
      int gj = jbase + jtl * 16 + l16;          // global col in [0, 384)
      int bb = gm / N_;
      int nn = gm - bb * N_;
      uint16_t bits = f32_to_bf16_bits(c[v]);
      if (gj < D_) {
        gK[(((size_t)bb * H_ + (gj >> 4)) * N_ + nn) * DK_ + (gj & 15)] = bits;
      } else if (gj < 2 * D_) {
        int j = gj - D_;
        gV[(((size_t)bb * H_ + (j >> 4)) * N_ + nn) * DK_ + (j & 15)] = bits;
      } else {
        lK[((size_t)bb * N_ + nn) * D_ + (gj - 2 * D_)] = bits;
      }
    }
  }
}

// ---------------------------------------------------------------------------
// fixed_ctx = mean(emb, axis=1) @ W_fixed       (one 128-thread block per b)
// ---------------------------------------------------------------------------
__global__ void __launch_bounds__(128)
fixed_ctx_kernel(const float* __restrict__ emb, const float* __restrict__ Wf,
                 float* __restrict__ fixedc) {
  const int b = blockIdx.x, tid = threadIdx.x;
  __shared__ float mean[D_];
  const float* eb = emb + (size_t)b * N_ * D_;
  float acc = 0.f;
  for (int n = 0; n < N_; ++n) acc += eb[(size_t)n * D_ + tid];
  mean[tid] = acc * (1.0f / N_);
  __syncthreads();
  float r = 0.f;
  for (int k = 0; k < D_; ++k) r += mean[k] * Wf[k * D_ + tid];
  fixedc[(size_t)b * D_ + tid] = r;
}

// ---------------------------------------------------------------------------
// q = fixed_ctx + concat(first_emb, prev_emb) @ W_step   (block per b, 128 thr)
// step==0 uses the learned placeholder context.
// ---------------------------------------------------------------------------
__global__ void __launch_bounds__(128)
query_kernel(const float* __restrict__ emb, const float* __restrict__ Ws,
             const float* __restrict__ Wph, const float* __restrict__ fixedc,
             const int* __restrict__ sel, float* __restrict__ q,
             int step, int S) {
  const int b = blockIdx.x, tid = threadIdx.x;
  __shared__ float ctx[2 * D_];
  if (step == 0) {
    ctx[tid]      = Wph[tid];
    ctx[tid + D_] = Wph[tid + D_];
  } else {
    int s0 = sel[b * S + 0];
    int sp = sel[b * S + step - 1];
    ctx[tid]      = emb[((size_t)b * N_ + s0) * D_ + tid];
    ctx[tid + D_] = emb[((size_t)b * N_ + sp) * D_ + tid];
  }
  __syncthreads();
  float acc = fixedc[(size_t)b * D_ + tid];
  for (int i = 0; i < 2 * D_; ++i) acc += ctx[i] * Ws[i * D_ + tid];
  q[(size_t)b * D_ + tid] = acc;
}

// ---------------------------------------------------------------------------
// Masked MHA over N=1000, dk=16.  One 256-thread block per (b,h).
// Two-pass softmax with scores cached in LDS; wave32 shfl reductions.
// ---------------------------------------------------------------------------
__global__ void __launch_bounds__(256)
attn_kernel(const uint16_t* __restrict__ gK, const uint16_t* __restrict__ gV,
            const float* __restrict__ q, const uint8_t* __restrict__ visited,
            float* __restrict__ heads) {
  const int bh = blockIdx.x;                 // b*H + h
  const int b  = bh >> 3;
  const int h  = bh & 7;
  const int tid = threadIdx.x;
  const int lane = tid & 31, wv = tid >> 5;

  __shared__ float qh[DK_];
  __shared__ float sc[N_];
  __shared__ float red[8];
  __shared__ float vred[8][DK_];

  if (tid < DK_) qh[tid] = q[(size_t)b * D_ + h * DK_ + tid];
  __syncthreads();
  float qr[DK_];
  #pragma unroll
  for (int k = 0; k < DK_; ++k) qr[k] = qh[k];

  const uint16_t* Kb = gK + (size_t)bh * N_ * DK_;
  const uint16_t* Vb = gV + (size_t)bh * N_ * DK_;
  const uint8_t*  vis = visited + (size_t)b * N_;

  float lmax = -INFINITY;
  for (int n = tid; n < N_; n += 256) {
    if (n + 256 < N_) __builtin_prefetch(Kb + (size_t)(n + 256) * DK_, 0, 1);
    float s = -INFINITY;
    if (!vis[n]) {
      const uint32_t* p = (const uint32_t*)(Kb + (size_t)n * DK_);
      float acc = 0.f;
      #pragma unroll
      for (int i = 0; i < 8; ++i) {
        uint32_t u = p[i];
        acc += __uint_as_float(u << 16)         * qr[2 * i];
        acc += __uint_as_float(u & 0xFFFF0000u) * qr[2 * i + 1];
      }
      s = acc * 0.25f;                          // 1/sqrt(16)
    }
    sc[n] = s;
    lmax = fmaxf(lmax, s);
  }
  #pragma unroll
  for (int off = 16; off; off >>= 1) lmax = fmaxf(lmax, __shfl_xor(lmax, off, 32));
  if (lane == 0) red[wv] = lmax;
  __syncthreads();
  float m = red[0];
  #pragma unroll
  for (int w = 1; w < 8; ++w) m = fmaxf(m, red[w]);
  __syncthreads();

  float lsum = 0.f;
  float vacc[DK_];
  #pragma unroll
  for (int k = 0; k < DK_; ++k) vacc[k] = 0.f;
  for (int n = tid; n < N_; n += 256) {
    float w = __expf(sc[n] - m);                // masked rows -> exp(-inf)=0
    if (w != 0.f) {
      lsum += w;
      const uint32_t* p = (const uint32_t*)(Vb + (size_t)n * DK_);
      #pragma unroll
      for (int i = 0; i < 8; ++i) {
        uint32_t u = p[i];
        vacc[2 * i]     += w * __uint_as_float(u << 16);
        vacc[2 * i + 1] += w * __uint_as_float(u & 0xFFFF0000u);
      }
    }
  }
  #pragma unroll
  for (int off = 16; off; off >>= 1) lsum += __shfl_xor(lsum, off, 32);
  #pragma unroll
  for (int k = 0; k < DK_; ++k) {
    #pragma unroll
    for (int off = 16; off; off >>= 1) vacc[k] += __shfl_xor(vacc[k], off, 32);
  }
  if (lane == 0) {
    red[wv] = lsum;
    #pragma unroll
    for (int k = 0; k < DK_; ++k) vred[wv][k] = vacc[k];
  }
  __syncthreads();
  if (tid < DK_) {
    float s = 0.f, hv = 0.f;
    #pragma unroll
    for (int w = 0; w < 8; ++w) { s += red[w]; hv += vred[w][tid]; }
    heads[(size_t)b * D_ + h * DK_ + tid] = hv / s;   // stored as (B, H*dk) = (B,D)
  }
}

// ---------------------------------------------------------------------------
// glimpse = heads @ W_out; logits vs lK; tanh clip; mask; log_softmax; argmax.
// Writes log_p[b,step,:], pi[b,step]; updates visited + sel.  Block per b.
// ---------------------------------------------------------------------------
__global__ void __launch_bounds__(256)
logits_kernel(const uint16_t* __restrict__ lK, const float* __restrict__ heads,
              const float* __restrict__ Wo, uint8_t* __restrict__ visited,
              int* __restrict__ sel, float* __restrict__ logp,
              float* __restrict__ pi, int step, int S) {
  const int b = blockIdx.x, tid = threadIdx.x;
  const int lane = tid & 31, wv = tid >> 5;
  __shared__ float hb[D_];
  __shared__ float g[D_];
  __shared__ float sc[N_];
  __shared__ float red[8];
  __shared__ int   ired[8];

  if (tid < D_) hb[tid] = heads[(size_t)b * D_ + tid];
  __syncthreads();
  if (tid < D_) {
    float acc = 0.f;
    for (int k = 0; k < D_; ++k) acc += hb[k] * Wo[k * D_ + tid];
    g[tid] = acc;
  }
  __syncthreads();

  const uint8_t*  vis = visited + (size_t)b * N_;
  const uint16_t* Lb  = lK + (size_t)b * N_ * D_;
  float lmax = -INFINITY, bv = -INFINITY;
  int bi = 0x7fffffff;
  for (int n = tid; n < N_; n += 256) {
    if (n + 256 < N_) __builtin_prefetch(Lb + (size_t)(n + 256) * D_, 0, 1);
    float s = -INFINITY;
    if (!vis[n]) {
      const uint32_t* p = (const uint32_t*)(Lb + (size_t)n * D_);
      float acc = 0.f;
      #pragma unroll 16
      for (int i = 0; i < 64; ++i) {
        uint32_t u = p[i];
        acc += __uint_as_float(u << 16)         * g[2 * i];
        acc += __uint_as_float(u & 0xFFFF0000u) * g[2 * i + 1];
      }
      s = 10.f * tanhf(acc * 0.08838834764831845f);   // clip, 1/sqrt(128)
    }
    sc[n] = s;
    lmax = fmaxf(lmax, s);
    if (s > bv || (s == bv && n < bi)) { bv = s; bi = n; }
  }
  // --- max reduce ---
  #pragma unroll
  for (int off = 16; off; off >>= 1) lmax = fmaxf(lmax, __shfl_xor(lmax, off, 32));
  if (lane == 0) red[wv] = lmax;
  __syncthreads();
  float m = red[0];
  #pragma unroll
  for (int w = 1; w < 8; ++w) m = fmaxf(m, red[w]);
  __syncthreads();
  // --- sum reduce ---
  float lsum = 0.f;
  for (int n = tid; n < N_; n += 256) lsum += __expf(sc[n] - m);
  #pragma unroll
  for (int off = 16; off; off >>= 1) lsum += __shfl_xor(lsum, off, 32);
  if (lane == 0) red[wv] = lsum;
  __syncthreads();
  float Z = 0.f;
  #pragma unroll
  for (int w = 0; w < 8; ++w) Z += red[w];
  const float logZ = m + logf(Z);
  __syncthreads();
  // --- write log-probs ---
  float* out = logp + ((size_t)b * S + step) * N_;
  for (int n = tid; n < N_; n += 256) out[n] = sc[n] - logZ;
  // --- argmax (first-max tie-break, matches jnp.argmax) ---
  #pragma unroll
  for (int off = 16; off; off >>= 1) {
    float ov = __shfl_xor(bv, off, 32);
    int   oi = __shfl_xor(bi, off, 32);
    if (ov > bv || (ov == bv && oi < bi)) { bv = ov; bi = oi; }
  }
  if (lane == 0) { red[wv] = bv; ired[wv] = bi; }
  __syncthreads();
  if (tid == 0) {
    float fb = red[0]; int fi = ired[0];
    #pragma unroll
    for (int w = 1; w < 8; ++w)
      if (red[w] > fb || (red[w] == fb && ired[w] < fi)) { fb = red[w]; fi = ired[w]; }
    sel[b * S + step] = fi;
    visited[(size_t)b * N_ + fi] = 1;
    pi[b * S + step] = (float)fi;
  }
}

// ---------------------------------------------------------------------------
extern "C" void kernel_launch(void* const* d_in, const int* in_sizes, int n_in,
                              void* d_out, int out_size, void* d_ws, size_t ws_size,
                              hipStream_t stream) {
  (void)in_sizes; (void)n_in; (void)ws_size;
  const float* emb = (const float*)d_in[0];
  const float* Wn  = (const float*)d_in[1];
  const float* Wf  = (const float*)d_in[2];
  const float* Ws  = (const float*)d_in[3];
  const float* Wo  = (const float*)d_in[4];
  const float* Wph = (const float*)d_in[5];
  // n_steps lives on-device (d_in[6]); derive S host-side from out_size:
  // out_size = B*S*N + B*S  ->  S = out_size / (B*(N+1))  (= 64).
  const int S = out_size / (B_ * (N_ + 1));

  char* ws = (char*)d_ws;
  size_t off = 0;
  auto alloc = [&](size_t bytes) {
    char* p = ws + off;
    off += (bytes + 255) & ~(size_t)255;
    return p;
  };
  uint16_t* gK     = (uint16_t*)alloc((size_t)B_ * H_ * N_ * DK_ * 2);
  uint16_t* gV     = (uint16_t*)alloc((size_t)B_ * H_ * N_ * DK_ * 2);
  uint16_t* lK     = (uint16_t*)alloc((size_t)B_ * N_ * D_ * 2);
  float*    fixedc = (float*)alloc((size_t)B_ * D_ * 4);
  float*    q      = (float*)alloc((size_t)B_ * D_ * 4);
  float*    heads  = (float*)alloc((size_t)B_ * D_ * 4);
  uint8_t*  visited = (uint8_t*)alloc((size_t)B_ * N_);
  int*      sel    = (int*)alloc((size_t)B_ * S * 4);

  hipMemsetAsync(visited, 0, (size_t)B_ * N_, stream);

  dim3 ggrid(B_ * N_ / 16, 2);                  // 8000 M-tiles x 2 column halves
  precompute_gemm<<<ggrid, 128, 0, stream>>>(emb, Wn, gK, gV, lK);
  fixed_ctx_kernel<<<B_, D_, 0, stream>>>(emb, Wf, fixedc);

  float* logp = (float*)d_out;
  float* pi   = (float*)d_out + (size_t)B_ * S * N_;

  for (int s = 0; s < S; ++s) {
    query_kernel<<<B_, D_, 0, stream>>>(emb, Ws, Wph, fixedc, sel, q, s, S);
    attn_kernel<<<B_ * H_, 256, 0, stream>>>(gK, gV, q, visited, heads);
    logits_kernel<<<B_, 256, 0, stream>>>(lK, heads, Wo, visited, sel, logp, pi, s, S);
  }
}
